// MultiHeadAttention_32255204393290
// MI455X (gfx1250) — matrix operand order repaired and checked by reading, not compiled, not run on hardware
//
#include <hip/hip_runtime.h>

// ---------------- problem constants ----------------
constexpr int B_ = 2;
constexpr int T_ = 4096;
constexpr int C_ = 768;
constexpr int H_ = 12;
constexpr int D_ = 64;          // head dim
constexpr int BT = B_ * T_;     // 8192 rows
constexpr int C3 = 3 * C_;      // 2304

typedef __attribute__((ext_vector_type(16))) _Float16 v16h;
typedef __attribute__((ext_vector_type(8)))  _Float16 v8h;
typedef __attribute__((ext_vector_type(8)))  float    v8f;

// ---------------- WMMA helpers (wave32, CDNA5) ----------------
// D = A(16x32 f16) * B(32x16 f16) + C(16x16 f32)
__device__ inline v8f wmma_f16(v16h a, v16h b, v8f c) {
    return __builtin_amdgcn_wmma_f32_16x16x32_f16(
        /*neg_a=*/false, a, /*neg_b=*/false, b,
        /*c_mod=*/(short)0, c, /*reuse_a=*/false, /*reuse_b=*/false);
}

// A-fragment: lane (l,hi) holds row row0+l; elements 0..7 = K k0+hi*8+0..7,
// elements 8..15 = K k0+16+hi*8+0..7   (16-bit A 16x32 layout, ISA 7.12.2)
__device__ inline v16h load_a_frag(const _Float16* __restrict__ base, int ld,
                                   int row0, int k0, int l, int hi) {
    const _Float16* p = base + (size_t)(row0 + l) * ld + k0 + hi * 8;
    v8h lo = *(const v8h*)(p);
    v8h hv = *(const v8h*)(p + 16);
    v16h a;
#pragma unroll
    for (int i = 0; i < 8; ++i) { a[i] = lo[i]; a[i + 8] = hv[i]; }
    return a;
}

// B-fragment from a [N][K] (transposed) matrix: lane (l,hi) holds column
// n0+l; elements 0..7 = K k0+hi*8+0..7, elements 8..15 = K k0+16+hi*8+0..7.
__device__ inline v16h load_b_frag(const _Float16* __restrict__ base, int ld,
                                   int n0, int k0, int l, int hi) {
    const _Float16* p = base + (size_t)(n0 + l) * ld + k0 + hi * 8;
    v8h lo = *(const v8h*)(p);
    v8h hv = *(const v8h*)(p + 16);
    v16h b;
#pragma unroll
    for (int i = 0; i < 8; ++i) { b[i] = lo[i]; b[i + 8] = hv[i]; }
    return b;
}

// ---------------- conversion / transpose kernels ----------------
__global__ void cvt_f32_f16(const float* __restrict__ src,
                            _Float16* __restrict__ dst, int n) {
    int i = blockIdx.x * blockDim.x + threadIdx.x;
    if (i < n) dst[i] = (_Float16)src[i];
}

// W:[K][N] f32 row-major  ->  Wt:[N][K] f16
__global__ void transpose_f32_f16(const float* __restrict__ W,
                                  _Float16* __restrict__ Wt, int K, int N) {
    int i = blockIdx.x * blockDim.x + threadIdx.x;
    if (i < K * N) {
        int n = i / K, k = i % K;
        Wt[i] = (_Float16)W[(size_t)k * N + n];
    }
}

// ---------------- QKV projection GEMM ----------------
// Xh:[BT][C] f16, Wth:[3C][C] f16 (transposed). One wave -> one 64x32 tile
// (4x2 fragment grid: 8 WMMAs per 32-wide k-step, 64 acc VGPRs, no spills).
// Scatter results into Q,K:[B,H,T,D] and Vt:[B,H,D,T] (f16).
__global__ void __launch_bounds__(256, 1) qkv_gemm(
        const _Float16* __restrict__ Xh,
        const _Float16* __restrict__ Wth,
        const float* __restrict__ b_attn,
        _Float16* __restrict__ Q,
        _Float16* __restrict__ K,
        _Float16* __restrict__ Vt) {
    const int lane = threadIdx.x & 31;
    const int l = lane & 15, hi = lane >> 4;
    const int wid = blockIdx.x * (blockDim.x >> 5) + (threadIdx.x >> 5);
    const int NT = C3 / 32;                    // 72 column tiles
    const int m0 = (wid / NT) * 64;
    const int n0 = (wid % NT) * 32;

    v8f acc[4][2];
#pragma unroll
    for (int mi = 0; mi < 4; ++mi)
#pragma unroll
        for (int nj = 0; nj < 2; ++nj) acc[mi][nj] = (v8f){};

    for (int kk = 0; kk < C_; kk += 32) {
        v16h a[4];
#pragma unroll
        for (int mi = 0; mi < 4; ++mi)
            a[mi] = load_a_frag(Xh, C_, m0 + mi * 16, kk, l, hi);
#pragma unroll
        for (int nj = 0; nj < 2; ++nj) {
            v16h b = load_b_frag(Wth, C_, n0 + nj * 16, kk, l, hi);
#pragma unroll
            for (int mi = 0; mi < 4; ++mi)
                acc[mi][nj] = wmma_f16(a[mi], b, acc[mi][nj]);
        }
    }

    // a 32-wide aligned column band lies entirely inside one of Q/K/V and
    // inside one head, so `which` and `h` are uniform per wave
#pragma unroll
    for (int nj = 0; nj < 2; ++nj) {
        const int n = n0 + nj * 16 + l;
        const float bias = b_attn[n];
        const int which = n / C_;              // 0:Q 1:K 2:V
        const int cm = n % C_;
        const int h = cm / D_, d = cm % D_;
#pragma unroll
        for (int mi = 0; mi < 4; ++mi) {
#pragma unroll
            for (int r = 0; r < 8; ++r) {
                int m = m0 + mi * 16 + r + 8 * hi;     // global token row
                int bb = m / T_, t = m % T_;
                _Float16 v = (_Float16)(acc[mi][nj][r] + bias);
                if (which == 0)
                    Q[(((size_t)bb * H_ + h) * T_ + t) * D_ + d] = v;
                else if (which == 1)
                    K[(((size_t)bb * H_ + h) * T_ + t) * D_ + d] = v;
                else
                    Vt[(((size_t)bb * H_ + h) * D_ + d) * T_ + t] = v;
            }
        }
    }
}

// ---------------- flash attention (causal, online softmax) ----------------
// One wave handles one (b,h,32-row q tile): 2 M-fragments so every K and V
// fragment is reused twice. KV streamed in 32-wide chunks (16 WMMAs/chunk).
__global__ void __launch_bounds__(128, 1) flash_attn(
        const _Float16* __restrict__ Q,
        const _Float16* __restrict__ K,
        const _Float16* __restrict__ Vt,
        _Float16* __restrict__ Y) {
    __shared__ _Float16 ptile[4][32][32];      // per-wave P staging tile
    const int lane = threadIdx.x & 31;
    const int l = lane & 15, hi = lane >> 4;
    const int w = threadIdx.x >> 5;
    const int wid = blockIdx.x * 4 + w;
    const int QT = T_ / 32;                    // 128 q tiles per (b,h)
    const int b = wid / (H_ * QT);
    const int rem = wid % (H_ * QT);
    const int h = rem / QT;
    const int qb = (rem % QT) * 32;

    const _Float16* Qb = Q + ((size_t)(b * H_ + h)) * T_ * D_;
    const _Float16* Kb = K + ((size_t)(b * H_ + h)) * T_ * D_;
    const _Float16* Vb = Vt + ((size_t)(b * H_ + h)) * D_ * T_;

    v16h aq[2][2];
#pragma unroll
    for (int mf = 0; mf < 2; ++mf) {
        aq[mf][0] = load_a_frag(Qb, D_, qb + mf * 16, 0, l, hi);
        aq[mf][1] = load_a_frag(Qb, D_, qb + mf * 16, 32, l, hi);
    }

    v8f acc[2][4];
    float m_i[2][8], l_i[2][8];
#pragma unroll
    for (int mf = 0; mf < 2; ++mf) {
#pragma unroll
        for (int j = 0; j < 4; ++j) acc[mf][j] = (v8f){};
#pragma unroll
        for (int r = 0; r < 8; ++r) { m_i[mf][r] = -1.0e30f; l_i[mf][r] = 0.0f; }
    }

    const float scale = 0.125f;                // 1/sqrt(64)
    const int nch = qb / 32 + 1;               // causal #32-wide KV chunks
    for (int ch = 0; ch < nch; ++ch) {
        const int kc = ch * 32;

        // ---- S = Q*K^T: two 16-col tiles; K frags dropped after use ----
        v8f s[2][2];                           // [mf][ntile]
#pragma unroll
        for (int nt = 0; nt < 2; ++nt) {
            v16h bk0 = load_b_frag(Kb, D_, kc + nt * 16, 0, l, hi);
            v16h bk1 = load_b_frag(Kb, D_, kc + nt * 16, 32, l, hi);
#pragma unroll
            for (int mf = 0; mf < 2; ++mf) {
                v8f t = {};
                t = wmma_f16(aq[mf][0], bk0, t);
                t = wmma_f16(aq[mf][1], bk1, t);
                s[mf][nt] = t;
            }
        }

        // ---- causal mask + online softmax (rows live in VGPR index r) ----
        const int c0 = kc + l, c1 = kc + 16 + l;
#pragma unroll
        for (int mf = 0; mf < 2; ++mf) {
#pragma unroll
            for (int r = 0; r < 8; ++r) {
                const int row = qb + mf * 16 + r + 8 * hi;
                float x0 = (c0 <= row) ? s[mf][0][r] * scale : -3.0e38f;
                float x1 = (c1 <= row) ? s[mf][1][r] * scale : -3.0e38f;
                float mx = fmaxf(x0, x1);
#pragma unroll
                for (int off = 1; off < 16; off <<= 1)
                    mx = fmaxf(mx, __shfl_xor(mx, off));
                const float mnew = fmaxf(m_i[mf][r], mx);
                const float f = __expf(m_i[mf][r] - mnew);
                const float p0 = __expf(x0 - mnew);
                const float p1 = __expf(x1 - mnew);
                float rs = p0 + p1;
#pragma unroll
                for (int off = 1; off < 16; off <<= 1)
                    rs += __shfl_xor(rs, off);
                l_i[mf][r] = l_i[mf][r] * f + rs;
                m_i[mf][r] = mnew;
#pragma unroll
                for (int j = 0; j < 4; ++j) acc[mf][j][r] *= f;
                // D-frag layout -> LDS tile (row = mf*16+r+8*hi, col = l/16+l)
                ptile[w][mf * 16 + r + 8 * hi][l] = (_Float16)p0;
                ptile[w][mf * 16 + r + 8 * hi][16 + l] = (_Float16)p1;
            }
        }

        // ---- reload P as A-fragments (per-wave LDS; DS ops in-order) ----
        v16h ap[2];
#pragma unroll
        for (int mf = 0; mf < 2; ++mf)
#pragma unroll
            for (int i = 0; i < 8; ++i) {
                ap[mf][i]     = ptile[w][mf * 16 + l][hi * 8 + i];
                ap[mf][i + 8] = ptile[w][mf * 16 + l][16 + hi * 8 + i];
            }

        // ---- acc += P * V  (V transposed: rows are d, contiguous t) ----
#pragma unroll
        for (int j = 0; j < 4; ++j) {
            v16h bv = load_b_frag(Vb, T_, j * 16, kc, l, hi);
            acc[0][j] = wmma_f16(ap[0], bv, acc[0][j]);
            acc[1][j] = wmma_f16(ap[1], bv, acc[1][j]);
        }
    }

    // ---- normalize and store y[b][t][h*D+d] as f16 ----
#pragma unroll
    for (int mf = 0; mf < 2; ++mf)
#pragma unroll
        for (int r = 0; r < 8; ++r) {
            const float inv = 1.0f / l_i[mf][r];
            const int t = qb + mf * 16 + r + 8 * hi;
            const size_t yrow = ((size_t)b * T_ + t) * C_ + h * D_;
#pragma unroll
            for (int j = 0; j < 4; ++j)
                Y[yrow + j * 16 + l] = (_Float16)(acc[mf][j][r] * inv);
        }
}

// ---------------- output projection GEMM ----------------
// Block = 8 waves arranged 4(M) x 2(N): block tile 256x64. The shared
// 64x64 B chunk is staged into LDS with CDNA5 async loads
// (global_load_async_to_lds_b128 + s_wait_asynccnt), then each wave reads
// B-fragments from LDS (cross-wave reuse). 64x32 tile per wave.
__global__ void __launch_bounds__(256, 1) proj_gemm(
        const _Float16* __restrict__ Yh,
        const _Float16* __restrict__ Wpt,
        const float* __restrict__ b_proj,
        float* __restrict__ out) {
    __shared__ _Float16 Bs[64][64];            // 8 KB staging tile
    const int tid = threadIdx.x;
    const int lane = tid & 31;
    const int l = lane & 15, hi = lane >> 4;
    const int w = tid >> 5;                    // 0..7
    const int wm = w & 3, wn = w >> 2;         // 4 M-waves x 2 N-waves
    const int NB = C_ / 64;                    // 12 column bands
    const int bm = (blockIdx.x / NB) * 256;
    const int bn = (blockIdx.x % NB) * 64;
    const int m0 = bm + wm * 64;
    const int n0loc = wn * 32;

    v8f acc[4][2];
#pragma unroll
    for (int mi = 0; mi < 4; ++mi)
#pragma unroll
        for (int nj = 0; nj < 2; ++nj) acc[mi][nj] = (v8f){};

    for (int kc = 0; kc < C_; kc += 64) {
        // ---- async-stage B[64 n][64 k] into LDS (2 b128 per thread) ----
#pragma unroll
        for (int i = 0; i < 2; ++i) {
            const int idx = tid * 2 + i;       // 0..511
            const int row = idx >> 3, seg = idx & 7;
            unsigned lds_off = (unsigned)(uintptr_t)&Bs[row][seg * 8];
            const _Float16* g = Wpt + (size_t)(bn + row) * C_ + kc + seg * 8;
            asm volatile("global_load_async_to_lds_b128 %0, %1, off"
                         :: "v"(lds_off),
                            "v"((unsigned long long)(uintptr_t)g)
                         : "memory");
        }
        asm volatile("s_wait_asynccnt 0x0" ::: "memory");
        __syncthreads();

#pragma unroll
        for (int ks = 0; ks < 2; ++ks) {
            const int kk = kc + ks * 32;
            v16h a[4];
#pragma unroll
            for (int mi = 0; mi < 4; ++mi)
                a[mi] = load_a_frag(Yh, C_, m0 + mi * 16, kk, l, hi);
#pragma unroll
            for (int nj = 0; nj < 2; ++nj) {
                const _Float16* pb = &Bs[n0loc + nj * 16 + l][ks * 32 + hi * 8];
                v16h b;
#pragma unroll
                for (int i = 0; i < 16; ++i) b[i] = pb[i < 8 ? i : i + 8];
#pragma unroll
                for (int mi = 0; mi < 4; ++mi)
                    acc[mi][nj] = wmma_f16(a[mi], b, acc[mi][nj]);
            }
        }
        __syncthreads();                       // before next stage overwrites
    }

#pragma unroll
    for (int nj = 0; nj < 2; ++nj) {
        const int n = bn + n0loc + nj * 16 + l;
        const float bias = b_proj[n];
#pragma unroll
        for (int mi = 0; mi < 4; ++mi)
#pragma unroll
            for (int r = 0; r < 8; ++r) {
                const int m = m0 + mi * 16 + r + 8 * hi;
                out[(size_t)m * C_ + n] = acc[mi][nj][r] + bias;
            }
    }
}

// ---------------- launcher ----------------
extern "C" void kernel_launch(void* const* d_in, const int* in_sizes, int n_in,
                              void* d_out, int out_size, void* d_ws, size_t ws_size,
                              hipStream_t stream) {
    const float* x      = (const float*)d_in[0];
    const float* w_attn = (const float*)d_in[1];
    const float* b_attn = (const float*)d_in[2];
    const float* w_proj = (const float*)d_in[3];
    const float* b_proj = (const float*)d_in[4];
    float* out = (float*)d_out;

    // workspace carve-up (all f16): ~68 MB total
    _Float16* p = (_Float16*)d_ws;
    _Float16* Xh  = p; p += (size_t)BT * C_;    // 8192*768
    _Float16* Wth = p; p += (size_t)C3 * C_;    // 2304*768 (transposed w_attn)
    _Float16* Wpt = p; p += (size_t)C_ * C_;    // 768*768  (transposed w_proj)
    _Float16* Qh  = p; p += (size_t)BT * C_;    // [B,H,T,D]
    _Float16* Kh  = p; p += (size_t)BT * C_;    // [B,H,T,D]
    _Float16* Vth = p; p += (size_t)BT * C_;    // [B,H,D,T]
    _Float16* Yh  = p; p += (size_t)BT * C_;    // attention output, f16

    // 1) convert / transpose to f16
    {
        int n = BT * C_;
        cvt_f32_f16<<<(n + 255) / 256, 256, 0, stream>>>(x, Xh, n);
    }
    {
        int n = C_ * C3;
        transpose_f32_f16<<<(n + 255) / 256, 256, 0, stream>>>(w_attn, Wth, C_, C3);
    }
    {
        int n = C_ * C_;
        transpose_f32_f16<<<(n + 255) / 256, 256, 0, stream>>>(w_proj, Wpt, C_, C_);
    }

    // 2) QKV projection: (8192/64)*(2304/32) = 9216 wave-tiles, 8 waves/block
    qkv_gemm<<<9216 / 8, 256, 0, stream>>>(Xh, Wth, b_attn, Qh, Kh, Vth);

    // 3) flash attention: B*H*(T/32) = 3072 waves, 4 waves/block
    flash_attn<<<3072 / 4, 128, 0, stream>>>(Qh, Kh, Vth, Yh);

    // 4) output projection: block tile 256x64 -> (8192/256)*(768/64) = 384 blocks
    proj_gemm<<<384, 256, 0, stream>>>(Yh, Wpt, b_proj, out);
}
